// HetGNN_14817637171202
// MI455X (gfx1250) — matrix operand order repaired and checked by reading, not compiled, not run on hardware
//
#include <hip/hip_runtime.h>

// ---------------------------------------------------------------------------
// HetGNN (2x HeteroGraphConv over 3 relations + 1-step BiLSTM) for gfx1250.
// GEMMs run on v_wmma_f32_16x16x32_f16 (wave32, one 16x16 tile per wave,
// K=128 as 4 WMMAs with all fragments preloaded so the load clause overlaps
// the WMMA chain). Edge aggregation is the bandwidth/atomic-bound part
// (~3.1 GB traffic -> ~135us floor @ 23.3 TB/s); the 192MB L2 holds the full
// 51MB feature buffers so scatter atomics stay on-chip.
// ---------------------------------------------------------------------------

typedef __attribute__((ext_vector_type(16))) _Float16 v16h;
typedef __attribute__((ext_vector_type(8)))  _Float16 v8h;
typedef __attribute__((ext_vector_type(8)))  float    v8f;

#define NREL 3
#define FDIM 128

// -------------------------------- utility ----------------------------------

__global__ void hg_zero_f32(float* __restrict__ p, long n) {
  long i = (long)blockIdx.x * blockDim.x + threadIdx.x;
  long stride = (long)gridDim.x * blockDim.x;
  for (; i < n; i += stride) p[i] = 0.0f;
}

__global__ void hg_to_f16(const float* __restrict__ s, _Float16* __restrict__ d, long n) {
  long i = (long)blockIdx.x * blockDim.x + threadIdx.x;
  if (i < n) d[i] = (_Float16)s[i];
}

__global__ void hg_to_f16_relu(const float* __restrict__ s, _Float16* __restrict__ d, long n) {
  long i = (long)blockIdx.x * blockDim.x + threadIdx.x;
  if (i < n) d[i] = (_Float16)fmaxf(s[i], 0.0f);
}

// W[r][k][j] (R x 128 x 128, row-major) -> Wt16[r][j][k]  (B^T layout for WMMA)
__global__ void hg_w_transpose_f16(const float* __restrict__ W, _Float16* __restrict__ Wt,
                                   int total) {
  int i = blockIdx.x * blockDim.x + threadIdx.x;
  if (i >= total) return;
  int r = i >> 14;            // /16384
  int rem = i & 16383;
  int k = rem >> 7, j = rem & 127;
  Wt[((size_t)r << 14) + ((size_t)j << 7) + k] = (_Float16)W[i];
}

// ------------------------------ graph kernels ------------------------------

__global__ void hg_degree(const int* __restrict__ src, const int* __restrict__ dst,
                          float* __restrict__ deg_out, float* __restrict__ deg_in,
                          int total, int E, int N) {
  int i = blockIdx.x * blockDim.x + threadIdx.x;
  if (i >= total) return;
  int r = i / E;
  atomicAdd(deg_out + (size_t)r * N + src[i], 1.0f);
  atomicAdd(deg_in  + (size_t)r * N + dst[i], 1.0f);
}

__global__ void hg_rsqrt_clip(float* __restrict__ d, long n) {
  long i = (long)blockIdx.x * blockDim.x + threadIdx.x;
  if (i < n) d[i] = rsqrtf(fmaxf(d[i], 1.0f));
}

// acc[dst] += y[src] * rsqrt(deg_out[src])
// One wave32 per edge; each lane owns 4 consecutive columns (float4 gather ->
// global_load_b128, then 4 global_atomic_add_f32). 4 edges per 128-thr block.
__global__ void hg_scatter(const float4* __restrict__ y4, float* __restrict__ acc,
                           const int* __restrict__ src, const int* __restrict__ dst,
                           const float* __restrict__ sout, int E) {
  int e = blockIdx.x * 4 + (threadIdx.x >> 5);
  if (e >= E) return;
  int lane = threadIdx.x & 31;
  int s = src[e], d = dst[e];
  float sc = sout[s];
  float4 v = y4[(size_t)s * 32 + lane];
  float* a = acc + (size_t)d * FDIM + lane * 4;
  atomicAdd(a + 0, v.x * sc);
  atomicAdd(a + 1, v.y * sc);
  atomicAdd(a + 2, v.z * sc);
  atomicAdd(a + 3, v.w * sc);
}

// hsum += (acc * rsqrt(deg_in)[:,None] + b) / 3
__global__ void hg_finalize(const float* __restrict__ acc, const float* __restrict__ sin,
                            const float* __restrict__ b, float* __restrict__ hsum, long n) {
  long i = (long)blockIdx.x * blockDim.x + threadIdx.x;
  if (i >= n) return;
  int node = (int)(i >> 7), c = (int)(i & 127);
  hsum[i] += (acc[i] * sin[node] + b[c]) * (1.0f / 3.0f);
}

// ------------------------------ WMMA GEMM ----------------------------------
// C[M,Ncols] = A16[M,128] @ Bt16[Ncols,128]^T   (f32 accum)
// One wave32 per 16x16 tile. Fragment layouts used here (contiguous k, same order for A and B):
//   A lane l (h=l>>4, m=l&15): halves 0..15 <- K = 16h .. 16h+15
//   B lane l (h=l>>4, n=l&15): halves 0..15 <- K = 16h .. 16h+15 (from Bt row)
//   D vgpr j, lane l:          C[m = 16*tm + j + 8h, n = 16*tn + (l&15)]
// All 4 K-chunks are preloaded so loads issue as one clause and the WMMA
// chain drains with staggered s_wait_loadcnt instead of full stalls.
__global__ __launch_bounds__(256)
void hg_gemm_wmma(const _Float16* __restrict__ A, const _Float16* __restrict__ Bt,
                  float* __restrict__ C, int mtiles, int ntiles) {
  int gid  = blockIdx.x * blockDim.x + threadIdx.x;
  int wave = gid >> 5;
  if (wave >= mtiles * ntiles) return;   // wave-uniform: EXEC stays all-ones for WMMA
  int lane = threadIdx.x & 31;
  int tm = wave / ntiles;
  int tn = wave - tm * ntiles;
  int h = lane >> 4, idx = lane & 15;

  const _Float16* arow = A  + (size_t)(tm * 16 + idx) * FDIM;
  const _Float16* brow = Bt + (size_t)(tn * 16 + idx) * FDIM;

  v16h a[4], b[4];
#pragma unroll
  for (int kc = 0; kc < 4; ++kc) {
    int kb = kc * 32;
    a[kc] = *(const v16h*)(arow + kb + 16 * h);                      // K = 16h..+15
    b[kc] = *(const v16h*)(brow + kb + 16 * h);                      // K = 16h..+15
  }

  v8f acc = {};
#pragma unroll
  for (int kc = 0; kc < 4; ++kc)
    acc = __builtin_amdgcn_wmma_f32_16x16x32_f16(
        /*neg_a=*/false, a[kc], /*neg_b=*/false, b[kc],
        /*c_mod=*/(short)0, acc, /*reuse_a=*/false, /*reuse_b=*/false);

  int Ncols = ntiles << 4;
  float* cb = C + (size_t)(tm * 16 + 8 * h) * Ncols + tn * 16 + idx;
#pragma unroll
  for (int j = 0; j < 8; ++j)
    cb[(size_t)j * Ncols] = acc[j];
}

// ------------------------------ LSTM gates ---------------------------------
// G[N,256] pre-activations (no bias yet); gate order i,f,g,o (f unused: c0=0).
__global__ void hg_lstm_gates(const float* __restrict__ G, const float* __restrict__ bih,
                              const float* __restrict__ bhh, float* __restrict__ out,
                              long total, int halfoff) {
  long i = (long)blockIdx.x * blockDim.x + threadIdx.x;
  if (i >= total) return;
  int n = (int)(i >> 6), j = (int)(i & 63);
  const float* g = G + (size_t)n * 256;
  float gi = g[j]       + bih[j]       + bhh[j];
  float gg = g[128 + j] + bih[128 + j] + bhh[128 + j];
  float go = g[192 + j] + bih[192 + j] + bhh[192 + j];
  float si = 1.0f / (1.0f + expf(-gi));
  float c  = si * tanhf(gg);
  float so = 1.0f / (1.0f + expf(-go));
  out[(size_t)n * FDIM + halfoff + j] = so * tanhf(c);
}

// ------------------------------- launcher ----------------------------------

static inline int ceil_div(long a, int b) { return (int)((a + b - 1) / b); }

extern "C" void kernel_launch(void* const* d_in, const int* in_sizes, int n_in,
                              void* d_out, int out_size, void* d_ws, size_t ws_size,
                              hipStream_t stream) {
  const float* x     = (const float*)d_in[0];
  const int*   src   = (const int*)  d_in[1];
  const int*   dst   = (const int*)  d_in[2];
  const float* W1    = (const float*)d_in[3];
  const float* b1    = (const float*)d_in[4];
  const float* W2    = (const float*)d_in[5];
  const float* b2    = (const float*)d_in[6];
  const float* Wih_f = (const float*)d_in[7];
  const float* bih_f = (const float*)d_in[9];
  const float* bhh_f = (const float*)d_in[10];
  const float* Wih_b = (const float*)d_in[11];
  const float* bih_b = (const float*)d_in[13];
  const float* bhh_b = (const float*)d_in[14];
  float* out = (float*)d_out;

  const int N  = in_sizes[0] / FDIM;       // 100000 (divisible by 16)
  const int RE = in_sizes[1];              // 3 * E
  const int E  = RE / NREL;
  const long NF = (long)N * FDIM;

  // ---- workspace layout (all 16B-aligned) ----
  float*    s_out = (float*)d_ws;                       // NREL*N  (rsqrt deg_out)
  float*    s_in  = s_out + (size_t)NREL * N;           // NREL*N  (rsqrt deg_in)
  _Float16* x16   = (_Float16*)(s_in + (size_t)NREL * N);          // N*128
  _Float16* W1t   = x16 + NF;                           // 3*128*128
  _Float16* W2t   = W1t + NREL * 16384;                 // 3*128*128
  _Float16* Wf16  = W2t + NREL * 16384;                 // 256*128
  _Float16* Wb16  = Wf16 + 32768;                       // 256*128
  float*    ybuf  = (float*)(Wb16 + 32768);             // N*128
  float*    accb  = ybuf + NF;                          // N*128
  float*    hsum  = accb + NF;                          // N*128
  float*    G     = ybuf;                               // N*256 reuses ybuf+accb

  const int T = 256;

  // ---- degrees -> rsqrt scales ----
  hg_zero_f32<<<2048, T, 0, stream>>>(s_out, (long)2 * NREL * N);
  hg_degree<<<ceil_div(RE, T), T, 0, stream>>>(src, dst, s_out, s_in, RE, E, N);
  hg_rsqrt_clip<<<ceil_div((long)2 * NREL * N, T), T, 0, stream>>>(s_out, (long)2 * NREL * N);

  // ---- weight prep (f16, B^T layout) ----
  hg_w_transpose_f16<<<ceil_div(NREL * 16384, T), T, 0, stream>>>(W1, W1t, NREL * 16384);
  hg_w_transpose_f16<<<ceil_div(NREL * 16384, T), T, 0, stream>>>(W2, W2t, NREL * 16384);
  hg_to_f16<<<ceil_div(32768, T), T, 0, stream>>>(Wih_f, Wf16, 32768);   // already [n][k]
  hg_to_f16<<<ceil_div(32768, T), T, 0, stream>>>(Wih_b, Wb16, 32768);

  const int mtiles = N / 16;
  const int nt128  = FDIM / 16;                  // 8
  const int gemm128_blocks = ceil_div((long)mtiles * nt128 * 32, T);
  const int nt256  = 256 / 16;                   // 16
  const int gemm256_blocks = ceil_div((long)mtiles * nt256 * 32, T);
  const int scat_blocks = ceil_div(E, 4);        // 4 edges (waves) per block

  // ---- layer 1 ----
  hg_to_f16<<<ceil_div(NF, T), T, 0, stream>>>(x, x16, NF);
  hg_zero_f32<<<2048, T, 0, stream>>>(hsum, NF);
  for (int r = 0; r < NREL; ++r) {
    hg_gemm_wmma<<<gemm128_blocks, T, 0, stream>>>(x16, W1t + (size_t)r * 16384,
                                                   ybuf, mtiles, nt128);
    hg_zero_f32<<<2048, T, 0, stream>>>(accb, NF);
    hg_scatter<<<scat_blocks, FDIM, 0, stream>>>((const float4*)ybuf, accb,
                                                 src + (size_t)r * E,
                                                 dst + (size_t)r * E,
                                                 s_out + (size_t)r * N, E);
    hg_finalize<<<ceil_div(NF, T), T, 0, stream>>>(accb, s_in + (size_t)r * N,
                                                   b1 + (size_t)r * FDIM, hsum, NF);
  }

  // ---- layer 2 (relu fused into f16 conversion of h1) ----
  hg_to_f16_relu<<<ceil_div(NF, T), T, 0, stream>>>(hsum, x16, NF);
  hg_zero_f32<<<2048, T, 0, stream>>>(hsum, NF);
  for (int r = 0; r < NREL; ++r) {
    hg_gemm_wmma<<<gemm128_blocks, T, 0, stream>>>(x16, W2t + (size_t)r * 16384,
                                                   ybuf, mtiles, nt128);
    hg_zero_f32<<<2048, T, 0, stream>>>(accb, NF);
    hg_scatter<<<scat_blocks, FDIM, 0, stream>>>((const float4*)ybuf, accb,
                                                 src + (size_t)r * E,
                                                 dst + (size_t)r * E,
                                                 s_out + (size_t)r * N, E);
    hg_finalize<<<ceil_div(NF, T), T, 0, stream>>>(accb, s_in + (size_t)r * N,
                                                   b2 + (size_t)r * FDIM, hsum, NF);
  }

  // ---- BiLSTM single step (h0=c0=0 => Whh drops out, f-gate drops out) ----
  hg_to_f16<<<ceil_div(NF, T), T, 0, stream>>>(hsum, x16, NF);   // h2 -> f16

  hg_gemm_wmma<<<gemm256_blocks, T, 0, stream>>>(x16, Wf16, G, mtiles, nt256);
  hg_lstm_gates<<<ceil_div((long)N * 64, T), T, 0, stream>>>(G, bih_f, bhh_f, out,
                                                             (long)N * 64, 0);
  hg_gemm_wmma<<<gemm256_blocks, T, 0, stream>>>(x16, Wb16, G, mtiles, nt256);
  hg_lstm_gates<<<ceil_div((long)N * 64, T), T, 0, stream>>>(G, bih_b, bhh_b, out,
                                                             (long)N * 64, 64);
}
